// BiLSTM_6493990551905
// MI455X (gfx1250) — matrix operand order repaired and checked by reading, not compiled, not run on hardware
//
#include <hip/hip_runtime.h>
#include <hip/hip_bf16.h>

typedef unsigned short u16;

constexpr int kS  = 8192;
constexpr int kLC = 16;
constexpr int kDW = 256;
constexpr int kHC = 128;
constexpr int kH2 = 256;
constexpr int kT  = 64;

typedef __attribute__((ext_vector_type(16))) __bf16 v16bf;
typedef __attribute__((ext_vector_type(8)))  float  v8f;

struct B32 { uint4 a, b; };

__device__ __forceinline__ v16bf ld_frag(const u16* p0, const u16* p1) {
  B32 r;
  r.a = *(const uint4*)p0;
  r.b = *(const uint4*)p1;
  return __builtin_bit_cast(v16bf, r);
}

__device__ __forceinline__ u16 f2bf(float x) {
  union { float f; unsigned u; } v; v.f = x;
  unsigned r = v.u + 0x7FFFu + ((v.u >> 16) & 1u);
  return (u16)(r >> 16);
}

__device__ __forceinline__ float sigm(float x) { return 1.0f / (1.0f + expf(-x)); }

// ---------------------------------------------------------------------------
// bf16 WMMA GEMM, register-blocked: OUT[M,N](f32) = A[M,K] @ W[N,K]^T + bias.
// Each wave computes a 32x64 tile (2 m-tiles x 4 n-tiles): per 32-wide k-tile
// it loads 2 A-fragments + 4 B-fragments (12 x b128) and issues 8 WMMAs,
// reusing A across N and B across M (1.5 loads/WMMA vs 4 unblocked).
// grid = (N/64, M/256), block = 256 (8 waves).
// ---------------------------------------------------------------------------
__global__ void k_gemm_bf16(const u16* __restrict__ A, const u16* __restrict__ W,
                            const float* __restrict__ bias, float* __restrict__ OUT,
                            int N, int K) {
  const int lane = threadIdx.x & 31;
  const int wave = threadIdx.x >> 5;
  const int mt2 = blockIdx.y * 8 + wave;   // 32-row strip index
  const int nt0 = blockIdx.x * 4;          // first 16-col tile
  const int r   = lane & 15;
  const int hh  = lane >> 4;

  v8f acc[2][4];
#pragma unroll
  for (int ntl = 0; ntl < 4; ++ntl) {
    const float bv = bias[(nt0 + ntl) * 16 + r];
    v8f cinit = { bv, bv, bv, bv, bv, bv, bv, bv };
    acc[0][ntl] = cinit;
    acc[1][ntl] = cinit;
  }

  const u16* pa0 = A + (size_t)(mt2 * 32 + r) * K + hh * 8;
  const u16* pa1 = pa0 + (size_t)16 * K;
  const u16* pw0 = W + (size_t)(nt0 * 16 + r) * K + hh * 8;

  for (int k0 = 0; k0 < K; k0 += 32) {
    v16bf af[2];
    af[0] = ld_frag(pa0 + k0, pa0 + k0 + 16);
    af[1] = ld_frag(pa1 + k0, pa1 + k0 + 16);
    v16bf bf[4];
#pragma unroll
    for (int ntl = 0; ntl < 4; ++ntl) {
      const u16* pw = pw0 + (size_t)(ntl * 16) * K + k0;
      bf[ntl] = ld_frag(pw, pw + 16);
    }
#pragma unroll
    for (int mtl = 0; mtl < 2; ++mtl)
#pragma unroll
      for (int ntl = 0; ntl < 4; ++ntl)
        acc[mtl][ntl] = __builtin_amdgcn_wmma_f32_16x16x32_bf16(
            false, af[mtl], false, bf[ntl], (short)0, acc[mtl][ntl], false, false);
  }

  // C/D layout: lanes 0-15 -> M = e, lanes 16-31 -> M = 8+e, col n = lane&15
#pragma unroll
  for (int mtl = 0; mtl < 2; ++mtl) {
#pragma unroll
    for (int ntl = 0; ntl < 4; ++ntl) {
      float* po = OUT + (size_t)(mt2 * 32 + mtl * 16 + hh * 8) * N + (nt0 + ntl) * 16 + r;
#pragma unroll
      for (int e = 0; e < 8; ++e) po[(size_t)e * N] = acc[mtl][ntl][e];
    }
  }
}

// ---------------------------------------------------------------------------
// small helpers
// ---------------------------------------------------------------------------
__global__ void k_cvt_bf16(const float* __restrict__ s, u16* __restrict__ d, int n) {
  int i = blockIdx.x * blockDim.x + threadIdx.x;
  if (i < n) d[i] = f2bf(s[i]);
}

// Wc[512,192] = [char_Wih(512x64) | char_Whh(512x128)]  (bf16)
__global__ void k_build_wc(const float* __restrict__ wih, const float* __restrict__ whh,
                           u16* __restrict__ d) {
  int i = blockIdx.x * blockDim.x + threadIdx.x;
  if (i >= 512 * 192) return;
  int n = i / 192, c = i % 192;
  float v = (c < 64) ? wih[n * 64 + c] : whh[n * 128 + (c - 64)];
  d[i] = f2bf(v);
}

// A_t[8192,192] = [char_emb[charsets[s,t]] (64) | h_{t-1} (128)]  (bf16)
__global__ void k_char_buildA(const u16* __restrict__ cemb, const int* __restrict__ charsets,
                              const u16* __restrict__ hst, u16* __restrict__ Abuf, int t) {
  int i = blockIdx.x * blockDim.x + threadIdx.x;
  if (i >= kS * 192) return;
  int s = i / 192, c = i % 192;
  u16 v;
  if (c < 64) v = cemb[charsets[s * kLC + t] * 64 + c];
  else        v = t ? hst[s * kHC + (c - 64)] : (u16)0;
  Abuf[i] = v;
}

// char LSTM cell, batched over words; capture h at t == len-1
__global__ void k_char_cell(const float* __restrict__ g, float* __restrict__ cst,
                            u16* __restrict__ hst, const int* __restrict__ lens,
                            float* __restrict__ cfeat, int t) {
  int i = blockIdx.x * blockDim.x + threadIdx.x;
  if (i >= kS * kHC) return;
  int s = i / kHC, j = i % kHC;
  const float* gr = g + (size_t)s * (4 * kHC);
  float gi = gr[j], gf = gr[kHC + j], gg = gr[2 * kHC + j], go = gr[3 * kHC + j];
  float c = t ? cst[i] : 0.0f;
  c = sigm(gf) * c + sigm(gi) * tanhf(gg);
  float h = sigm(go) * tanhf(c);
  cst[i] = c;
  hst[i] = f2bf(h);
  if (t == lens[s] - 1) cfeat[i] = h;
}

// embeds[8192,384] = [char_feats(128) | word_emb[sentence](256)]  (bf16)
__global__ void k_build_embeds(const float* __restrict__ cfeat, const float* __restrict__ wemb,
                               const int* __restrict__ sent, u16* __restrict__ d) {
  int i = blockIdx.x * blockDim.x + threadIdx.x;
  if (i >= kS * 384) return;
  int s = i / 384, c = i % 384;
  float v = (c < kHC) ? cfeat[s * kHC + c]
                      : wemb[(size_t)sent[s] * kDW + (c - kHC)];
  d[i] = f2bf(v);
}

// ---------------------------------------------------------------------------
// Persistent word BiLSTM recurrence. grid=2 (fw,bw), block=1024 (32 waves).
// gates(1024) = Whh(1024x256) @ h  via WMMA with h replicated across B cols.
// Whh bf16 = 512KB: k-tiles 0..3 live in registers (A-fragments),
// k-tiles 4..7 live in LDS pre-swizzled into fragment layout (256KB).
// ---------------------------------------------------------------------------
__global__ void __launch_bounds__(1024, 1)
k_word_lstm(const u16* __restrict__ whh_fw, const u16* __restrict__ whh_bw,
            const float* __restrict__ P_fw, const float* __restrict__ P_bw,
            float* __restrict__ hf, float* __restrict__ hb) {
  extern __shared__ unsigned char smem_raw[];
  u16*   wlds = (u16*)smem_raw;                        // 262144 B: 256 frags x 32 lanes x 32B
  u16*   sh_h = (u16*)(smem_raw + 262144);             // 512 B
  float* sh_g = (float*)(smem_raw + 262144 + 512);     // 4096 B

  const int tid  = threadIdx.x;
  const int lane = tid & 31;
  const int wave = tid >> 5;
  const int r    = lane & 15;
  const int hh   = lane >> 4;
  const int dir  = blockIdx.x;

  const u16*   whh  = dir ? whh_bw : whh_fw;
  const float* P    = dir ? P_bw : P_fw;
  float*       Hout = dir ? hb : hf;

  // register-resident A fragments: mt = 2*wave+{0,1}, kt = 0..3
  v16bf areg[2][4];
#pragma unroll
  for (int mtl = 0; mtl < 2; ++mtl) {
    int mt = wave * 2 + mtl;
    const u16* base = whh + (size_t)(mt * 16 + r) * 256 + hh * 8;
#pragma unroll
    for (int kt = 0; kt < 4; ++kt)
      areg[mtl][kt] = ld_frag(base + kt * 32, base + kt * 32 + 16);
  }

  // LDS-resident fragments for kt = 4..7, pre-swizzled to per-lane 32B chunks
  for (int cid = tid; cid < 16384; cid += 1024) {
    int rr = cid & 1, ln = (cid >> 1) & 31, kt = 4 + ((cid >> 6) & 3), mt = cid >> 8;
    int r2 = ln & 15, h2 = ln >> 4;
    const uint4* src = (const uint4*)(whh + (size_t)(mt * 16 + r2) * 256 +
                                      kt * 32 + h2 * 8 + rr * 16);
    uint4* dst = (uint4*)(wlds + (mt * 4 + (kt - 4)) * 512 + ln * 16 + rr * 8);
    *dst = *src;
  }
  if (tid < kH2) sh_h[tid] = 0;
  __syncthreads();

  float creg = 0.0f;
  for (int step = 0; step < kS; ++step) {
    const int t = dir ? (kS - 1 - step) : step;

    // B fragments: B[k][n] = h[k] for all n (replicated); element e -> K = kt*32 + half*8 + e (e<8), kt*32 + 16 + half*8 + (e-8) (e>=8)
    v16bf breg[8];
#pragma unroll
    for (int kt = 0; kt < 8; ++kt) {
      const u16* ph = sh_h + kt * 32 + hh * 8;
      breg[kt] = ld_frag(ph, ph + 16);
    }

#pragma unroll
    for (int mtl = 0; mtl < 2; ++mtl) {
      const int mt = wave * 2 + mtl;
      v8f acc = { 0, 0, 0, 0, 0, 0, 0, 0 };
#pragma unroll
      for (int kt = 0; kt < 4; ++kt)
        acc = __builtin_amdgcn_wmma_f32_16x16x32_bf16(false, areg[mtl][kt], false,
                                                      breg[kt], (short)0, acc, false, false);
#pragma unroll
      for (int kt = 4; kt < 8; ++kt) {
        const uint4* p = (const uint4*)(wlds + (mt * 4 + (kt - 4)) * 512 + lane * 16);
        B32 rb; rb.a = p[0]; rb.b = p[1];
        v16bf lf = __builtin_bit_cast(v16bf, rb);
        acc = __builtin_amdgcn_wmma_f32_16x16x32_bf16(false, lf, false,
                                                      breg[kt], (short)0, acc, false, false);
      }
      if (r == 0) {  // column n==0 holds the matvec result
        float* pg = sh_g + mt * 16 + hh * 8;
#pragma unroll
        for (int e = 0; e < 8; ++e) pg[e] = acc[e];
      }
    }
    __syncthreads();

    if (tid < kH2) {
      const float* pp = P + (size_t)t * 1024;
      float gi = sh_g[tid]           + pp[tid];
      float gf = sh_g[kH2 + tid]     + pp[kH2 + tid];
      float gg = sh_g[2 * kH2 + tid] + pp[2 * kH2 + tid];
      float go = sh_g[3 * kH2 + tid] + pp[3 * kH2 + tid];
      creg = sigm(gf) * creg + sigm(gi) * tanhf(gg);
      float h = sigm(go) * tanhf(creg);
      Hout[(size_t)t * kH2 + tid] = h;
      sh_h[tid] = f2bf(h);
      int tn = dir ? (t - 1) : (t + 1);
      if (tn >= 0 && tn < kS) __builtin_prefetch(P + (size_t)tn * 1024 + tid, 0, 0);
    }
    __syncthreads();
  }
}

// lstm_out[8192,512] = [hf | hb]  (bf16)
__global__ void k_pack_out(const float* __restrict__ hf, const float* __restrict__ hb,
                           u16* __restrict__ d) {
  int i = blockIdx.x * blockDim.x + threadIdx.x;
  if (i >= kS * 512) return;
  int s = i / 512, c = i % 512;
  float v = (c < kH2) ? hf[s * kH2 + c] : hb[s * kH2 + (c - kH2)];
  d[i] = f2bf(v);
}

__global__ void k_logsoftmax(const float* __restrict__ logits, float* __restrict__ out) {
  int s = blockIdx.x * blockDim.x + threadIdx.x;
  if (s >= kS) return;
  const float* p = logits + (size_t)s * kT;
  float m = -1e30f;
  for (int j = 0; j < kT; ++j) m = fmaxf(m, p[j]);
  float sum = 0.0f;
  for (int j = 0; j < kT; ++j) sum += expf(p[j] - m);
  float lse = m + logf(sum);
  float* o = out + (size_t)s * kT;
  for (int j = 0; j < kT; ++j) o[j] = p[j] - lse;
}

// ---------------------------------------------------------------------------
extern "C" void kernel_launch(void* const* d_in, const int* in_sizes, int n_in,
                              void* d_out, int out_size, void* d_ws, size_t ws_size,
                              hipStream_t stream) {
  const int*   sentence = (const int*)d_in[0];
  const int*   charsets = (const int*)d_in[1];
  const int*   lens     = (const int*)d_in[2];
  const float* word_emb = (const float*)d_in[3];
  const float* char_emb = (const float*)d_in[4];
  const float* char_Wih = (const float*)d_in[5];
  const float* char_Whh = (const float*)d_in[6];
  const float* char_b   = (const float*)d_in[7];
  const float* fw_Wih   = (const float*)d_in[8];
  const float* fw_Whh   = (const float*)d_in[9];
  const float* fw_b     = (const float*)d_in[10];
  const float* bw_Wih   = (const float*)d_in[11];
  const float* bw_Whh   = (const float*)d_in[12];
  const float* bw_b     = (const float*)d_in[13];
  const float* out_W    = (const float*)d_in[14];
  const float* out_b    = (const float*)d_in[15];

  unsigned char* w = (unsigned char*)d_ws;
  auto alloc = [&](size_t bytes) -> void* {
    void* p = (void*)w;
    w += (bytes + 255) & ~(size_t)255;
    return p;
  };
  u16*   wc_bf    = (u16*)alloc((size_t)512 * 192 * 2);
  u16*   fwWih_bf = (u16*)alloc((size_t)1024 * 384 * 2);
  u16*   bwWih_bf = (u16*)alloc((size_t)1024 * 384 * 2);
  u16*   fwWhh_bf = (u16*)alloc((size_t)1024 * 256 * 2);
  u16*   bwWhh_bf = (u16*)alloc((size_t)1024 * 256 * 2);
  u16*   outW_bf  = (u16*)alloc((size_t)64 * 512 * 2);
  u16*   cemb_bf  = (u16*)alloc((size_t)128 * 64 * 2);
  u16*   Abuf     = (u16*)alloc((size_t)kS * 192 * 2);
  float* gates    = (float*)alloc((size_t)kS * 512 * 4);
  u16*   hstate   = (u16*)alloc((size_t)kS * kHC * 2);
  float* cstate   = (float*)alloc((size_t)kS * kHC * 4);
  float* cfeat    = (float*)alloc((size_t)kS * kHC * 4);
  u16*   embeds   = (u16*)alloc((size_t)kS * 384 * 2);
  float* P_fw     = (float*)alloc((size_t)kS * 1024 * 4);
  float* P_bw     = (float*)alloc((size_t)kS * 1024 * 4);
  float* hf       = (float*)alloc((size_t)kS * kH2 * 4);
  float* hb       = (float*)alloc((size_t)kS * kH2 * 4);
  u16*   lstm_bf  = (u16*)alloc((size_t)kS * 512 * 2);
  float* logits   = (float*)alloc((size_t)kS * kT * 4);

  const int TB = 256;
  // weight conversion to bf16
  k_build_wc<<<(512 * 192 + TB - 1) / TB, TB, 0, stream>>>(char_Wih, char_Whh, wc_bf);
  k_cvt_bf16<<<(1024 * 384 + TB - 1) / TB, TB, 0, stream>>>(fw_Wih, fwWih_bf, 1024 * 384);
  k_cvt_bf16<<<(1024 * 384 + TB - 1) / TB, TB, 0, stream>>>(bw_Wih, bwWih_bf, 1024 * 384);
  k_cvt_bf16<<<(1024 * 256 + TB - 1) / TB, TB, 0, stream>>>(fw_Whh, fwWhh_bf, 1024 * 256);
  k_cvt_bf16<<<(1024 * 256 + TB - 1) / TB, TB, 0, stream>>>(bw_Whh, bwWhh_bf, 1024 * 256);
  k_cvt_bf16<<<(64 * 512 + TB - 1) / TB, TB, 0, stream>>>(out_W, outW_bf, 64 * 512);
  k_cvt_bf16<<<(128 * 64 + TB - 1) / TB, TB, 0, stream>>>(char_emb, cemb_bf, 128 * 64);

  // char LSTM: 16 fused steps, each = gather + WMMA GEMM + cell
  for (int t = 0; t < kLC; ++t) {
    k_char_buildA<<<(kS * 192 + TB - 1) / TB, TB, 0, stream>>>(cemb_bf, charsets, hstate, Abuf, t);
    k_gemm_bf16<<<dim3(512 / 64, kS / 256), TB, 0, stream>>>(Abuf, wc_bf, char_b, gates, 512, 192);
    k_char_cell<<<(kS * kHC + TB - 1) / TB, TB, 0, stream>>>(gates, cstate, hstate, lens, cfeat, t);
  }

  // word-level embeddings and input projections
  k_build_embeds<<<(kS * 384 + TB - 1) / TB, TB, 0, stream>>>(cfeat, word_emb, sentence, embeds);
  k_gemm_bf16<<<dim3(1024 / 64, kS / 256), TB, 0, stream>>>(embeds, fwWih_bf, fw_b, P_fw, 1024, 384);
  k_gemm_bf16<<<dim3(1024 / 64, kS / 256), TB, 0, stream>>>(embeds, bwWih_bf, bw_b, P_bw, 1024, 384);

  // sequential recurrences (fw + bw persistent workgroups, WMMA matvec)
  const int lstm_smem = 262144 + 512 + 4096;
  hipFuncSetAttribute((const void*)k_word_lstm,
                      hipFuncAttributeMaxDynamicSharedMemorySize, lstm_smem);
  k_word_lstm<<<2, 1024, lstm_smem, stream>>>(fwWhh_bf, bwWhh_bf, P_fw, P_bw, hf, hb);

  // output projection + log_softmax
  k_pack_out<<<(kS * 512 + TB - 1) / TB, TB, 0, stream>>>(hf, hb, lstm_bf);
  k_gemm_bf16<<<dim3(kT / 64, kS / 256), TB, 0, stream>>>(lstm_bf, outW_bf, out_b, logits, kT, 512);
  k_logsoftmax<<<(kS + TB - 1) / TB, TB, 0, stream>>>(logits, (float*)d_out);
}
